// h2o_GAT_39178691674931
// MI455X (gfx1250) — matrix operand order repaired and checked by reading, not compiled, not run on hardware
//
#include <hip/hip_runtime.h>
#include <hip/hip_bf16.h>

#define BB 8
#define NH 512
#define NO 512
#define NN 1024
#define DD 128
#define HEADS 4
#define K2 (HEADS * DD)   // 512
#define ALPHA 0.2f

typedef __attribute__((ext_vector_type(16))) _Float16 v16h;
typedef __attribute__((ext_vector_type(8)))  float    v8f;

union Frag16 { v16h v; uint4 q[2]; };
union Pack8  { _Float16 h[8]; uint4 q; };

// ---------------- WMMA fragment loaders (wave32, 16-bit layouts per ISA 7.12.2) ---
// A (MxK row-major): lane L -> row = base+(L&15); K runs [k0,k0+8) and [k0+16,k0+24), k0 = 8*(L>>4)
__device__ inline v16h load_a16(const _Float16* base, int ld, int row, int kk, int lane) {
  Frag16 f;
  int r  = row + (lane & 15);
  int k0 = kk + ((lane >> 4) << 3);
  const _Float16* p = base + (size_t)r * ld + k0;
  f.q[0] = *(const uint4*)p;
  f.q[1] = *(const uint4*)(p + 16);
  return f.v;
}
// B (KxN) supplied as B^T row-major [N][K]: lane L -> col = base+(L&15); K runs [k0,k0+8) and [k0+16,k0+24), k0 = 8*(L>>4)
__device__ inline v16h load_b16(const _Float16* baseT, int ld, int col, int kk, int lane) {
  Frag16 f;
  int c  = col + (lane & 15);
  int k0 = kk + ((lane >> 4) << 3);
  const _Float16* p = baseT + (size_t)c * ld + k0;
  f.q[0] = *(const uint4*)p;
  f.q[1] = *(const uint4*)(p + 16);
  return f.v;
}

// ---- 2x2 register-blocked GEMM core: D(32x32) += A(32xK) * Bt(32xK)^T ------------
struct Acc4 { v8f c00, c01, c10, c11; };

__device__ inline void gemm_block_2x2(const _Float16* A, int lda, int row0,
                                      const _Float16* Bt, int ldb, int col0,
                                      int K, int lane, Acc4& acc) {
  for (int kk = 0; kk < K; kk += 32) {
    v16h a0 = load_a16(A, lda, row0,      kk, lane);
    v16h a1 = load_a16(A, lda, row0 + 16, kk, lane);
    v16h b0 = load_b16(Bt, ldb, col0,      kk, lane);
    v16h b1 = load_b16(Bt, ldb, col0 + 16, kk, lane);
    acc.c00 = __builtin_amdgcn_wmma_f32_16x16x32_f16(false, a0, false, b0, (short)0, acc.c00, false, false);
    acc.c01 = __builtin_amdgcn_wmma_f32_16x16x32_f16(false, a0, false, b1, (short)0, acc.c01, false, false);
    acc.c10 = __builtin_amdgcn_wmma_f32_16x16x32_f16(false, a1, false, b0, (short)0, acc.c10, false, false);
    acc.c11 = __builtin_amdgcn_wmma_f32_16x16x32_f16(false, a1, false, b1, (short)0, acc.c11, false, false);
  }
}

// Store one 16x16 f32 acc tile transposed as f16: dstT[e][n], e = D-col, n = D-row.
__device__ inline void store_projT(_Float16* dstT, int ldT, int e_base, int n_base,
                                   int lane, const v8f& acc) {
  int e  = e_base + (lane & 15);
  int n0 = n_base + ((lane >> 4) << 3);
  Pack8 t;
  #pragma unroll
  for (int v = 0; v < 8; ++v) t.h[v] = (_Float16)acc[v];
  *(uint4*)(dstT + (size_t)e * ldT + n0) = t.q;
}

// ---------------- K0: pack x to f16, transpose+pack weights ----------------------
__global__ void k0_prep(const float* __restrict__ h_states, const float* __restrict__ o_states,
                        const float* __restrict__ W_heads,  const float* __restrict__ W_out,
                        _Float16* __restrict__ xh, _Float16* __restrict__ WtH,
                        _Float16* __restrict__ Wt2) {
  int tid = blockIdx.x * blockDim.x + threadIdx.x;
  const int XN  = BB * NN * DD;
  const int WN  = HEADS * DD * DD;
  const int W2N = DD * K2;
  if (tid < XN) {
    int d = tid % DD;
    int n = (tid / DD) % NN;
    int b = tid / (DD * NN);
    float v = (n < NH) ? h_states[(size_t)(b * NH + n) * DD + d]
                       : o_states[(size_t)(b * NO + (n - NH)) * DD + d];
    xh[tid] = (_Float16)v;
  } else if (tid < XN + WN) {
    int t = tid - XN;
    int d = t % DD;
    int e = (t / DD) % DD;
    int h = t / (DD * DD);
    WtH[((size_t)h * DD + e) * DD + d] = (_Float16)W_heads[((size_t)h * DD + d) * DD + e];
  } else if (tid < XN + WN + W2N) {
    int t = tid - XN - WN;
    int k = t % K2;
    int e = t / K2;
    Wt2[(size_t)e * K2 + k] = (_Float16)W_out[(size_t)k * DD + e];
  }
}

// ---------------- K1: layer-1 projection, h1T[bh][e][n] = (x @ W_h)^T -------------
// 2x2 blocked: each wave computes 32 n-rows x 32 e-cols.
__global__ void k1_proj1(const _Float16* __restrict__ xh, const _Float16* __restrict__ WtH,
                         _Float16* __restrict__ h1T) {
  int lane = threadIdx.x & 31;
  int wave = (blockIdx.x * blockDim.x + threadIdx.x) >> 5;
  int bh = wave >> 7;              // 128 tiles of 32x32 per (b,head)
  int r  = wave & 127;
  int row0 = (r >> 2) << 5;        // n base (32 of 1024)
  int col0 = (r & 3) << 5;         // e base (32 of 128)
  int b = bh >> 2, hd = bh & 3;
  const _Float16* A  = xh  + (size_t)b  * NN * DD;
  const _Float16* Bt = WtH + (size_t)hd * DD * DD;
  Acc4 acc = {};
  gemm_block_2x2(A, DD, row0, Bt, DD, col0, DD, lane, acc);
  _Float16* dstT = h1T + (size_t)bh * DD * NN;
  store_projT(dstT, NN, col0,      row0,      lane, acc.c00);
  store_projT(dstT, NN, col0 + 16, row0,      lane, acc.c01);
  store_projT(dstT, NN, col0,      row0 + 16, lane, acc.c10);
  store_projT(dstT, NN, col0 + 16, row0 + 16, lane, acc.c11);
}

// ---------------- K2: f1/f2 matvecs, layer 1 --------------------------------------
__global__ void k2_fvec1(const _Float16* __restrict__ h1T, const float* __restrict__ a_heads,
                         float* __restrict__ f1, float* __restrict__ f2) {
  int tid = blockIdx.x * blockDim.x + threadIdx.x;
  if (tid >= BB * HEADS * NN) return;
  int n  = tid % NN;
  int bh = tid / NN;
  int hd = bh & 3;
  const _Float16* hp = h1T + (size_t)bh * DD * NN + n;
  const float* a1 = a_heads + hd * 2 * DD;
  const float* a2 = a1 + DD;
  float s1 = 0.f, s2 = 0.f;
  for (int e = 0; e < DD; ++e) {
    float hv = (float)hp[(size_t)e * NN];
    s1 += hv * a1[e];
    s2 += hv * a2[e];
  }
  f1[tid] = s1; f2[tid] = s2;
}

// ---------------- K3: masked softmax rows, layer 1 (structural mask) --------------
__global__ void k3_att1(const float* __restrict__ f1, const float* __restrict__ f2,
                        _Float16* __restrict__ att, float* __restrict__ att_self) {
  __shared__ float red[128];
  int row = blockIdx.x;            // bh*NN + n
  int t = threadIdx.x;
  int n  = row % NN;
  int bh = row / NN;
  bool human = n < NH;
  int opp0 = human ? NH : 0;
  float f1v = f1[(size_t)bh * NN + n];
  const float* f2r = f2 + (size_t)bh * NN + opp0;
  float s[4];
  float lmax = -3.0e38f;
  #pragma unroll
  for (int i = 0; i < 4; ++i) {
    float x = f1v + f2r[t + i * 128];
    x = x > 0.f ? x : ALPHA * x;
    s[i] = x;
    lmax = fmaxf(lmax, x);
  }
  float sself = 0.f;
  if (human) {
    float x = f1v + f2[(size_t)bh * NN + n];
    sself = x > 0.f ? x : ALPHA * x;
    lmax = fmaxf(lmax, sself);
  }
  red[t] = lmax; __syncthreads();
  for (int off = 64; off > 0; off >>= 1) { if (t < off) red[t] = fmaxf(red[t], red[t + off]); __syncthreads(); }
  float M = red[0]; __syncthreads();
  float lsum = 0.f;
  #pragma unroll
  for (int i = 0; i < 4; ++i) lsum += __expf(s[i] - M);
  if (human && t == 0) lsum += __expf(sself - M);
  red[t] = lsum; __syncthreads();
  for (int off = 64; off > 0; off >>= 1) { if (t < off) red[t] += red[t + off]; __syncthreads(); }
  float inv = 1.f / red[0];
  _Float16* arow = att + (size_t)row * 512;
  #pragma unroll
  for (int i = 0; i < 4; ++i) arow[t + i * 128] = (_Float16)(__expf(s[i] - M) * inv);
  if (t == 0) att_self[row] = human ? __expf(sself - M) * inv : 0.f;
}

// ---- K4 epilogue: +self (humans), elu, pack 8xf16, store into x2 -----------------
__device__ inline void store_x2_tile(const v8f& acc, const _Float16* hT,
                                     const float* as_bh, _Float16* x2_b, int hd,
                                     int n, int e0, bool human) {
  float aself = human ? as_bh[n] : 0.f;
  Pack8 t;
  #pragma unroll
  for (int v = 0; v < 8; ++v) {
    float val = acc[v];
    if (human) val += aself * (float)hT[(size_t)(e0 + v) * NN + n];
    val = val > 0.f ? val : (__expf(val) - 1.f);           // elu
    t.h[v] = (_Float16)val;
  }
  *(uint4*)(x2_b + (size_t)n * K2 + hd * DD + e0) = t.q;
}

// ---------------- K4: AV (layer 1) as out1^T = h1T x att^T; 2x2 blocked -----------
__global__ void k4_av1(const _Float16* __restrict__ h1T, const _Float16* __restrict__ att,
                       const float* __restrict__ att_self, _Float16* __restrict__ x2) {
  int lane = threadIdx.x & 31;
  int wave = (blockIdx.x * blockDim.x + threadIdx.x) >> 5;
  int bh = wave >> 7;              // 128 tiles of 32x32 per (b,head)
  int r  = wave & 127;
  int col0 = (r >> 2) << 5;        // n base (32 of 1024)
  int row0 = (r & 3) << 5;         // e base (32 of 128)
  int b = bh >> 2, hd = bh & 3;
  bool human = col0 < NH;
  int opp0 = human ? NH : 0;
  const _Float16* hT = h1T + (size_t)bh * DD * NN;
  const _Float16* A  = hT + opp0;                          // rows e, K = neighbor m
  const _Float16* Bt = att + (size_t)bh * NN * 512;        // B^T: att rows [n][m]
  Acc4 acc = {};
  gemm_block_2x2(A, NN, row0, Bt, 512, col0, 512, lane, acc);
  const float* as_bh = att_self + (size_t)bh * NN;
  _Float16* x2_b = x2 + (size_t)b * NN * K2;
  int nlo = col0 + (lane & 15), nhi = nlo + 16;
  int elo = row0 + ((lane >> 4) << 3), ehi = elo + 16;
  store_x2_tile(acc.c00, hT, as_bh, x2_b, hd, nlo, elo, human);
  store_x2_tile(acc.c01, hT, as_bh, x2_b, hd, nhi, elo, human);
  store_x2_tile(acc.c10, hT, as_bh, x2_b, hd, nlo, ehi, human);
  store_x2_tile(acc.c11, hT, as_bh, x2_b, hd, nhi, ehi, human);
}

// ---------------- K5: layer-2 projection, h2T[b][e][n] = (x2 @ W_out)^T -----------
__global__ void k5_proj2(const _Float16* __restrict__ x2, const _Float16* __restrict__ Wt2,
                         _Float16* __restrict__ h2T) {
  int lane = threadIdx.x & 31;
  int wave = (blockIdx.x * blockDim.x + threadIdx.x) >> 5;
  int b = wave >> 7;               // 128 tiles of 32x32 per graph
  int r = wave & 127;
  int row0 = (r >> 2) << 5;        // n base
  int col0 = (r & 3) << 5;         // e base
  const _Float16* A = x2 + (size_t)b * NN * K2;
  Acc4 acc = {};
  gemm_block_2x2(A, K2, row0, Wt2, K2, col0, K2, lane, acc);
  _Float16* dstT = h2T + (size_t)b * DD * NN;
  store_projT(dstT, NN, col0,      row0,      lane, acc.c00);
  store_projT(dstT, NN, col0 + 16, row0,      lane, acc.c01);
  store_projT(dstT, NN, col0,      row0 + 16, lane, acc.c10);
  store_projT(dstT, NN, col0 + 16, row0 + 16, lane, acc.c11);
}

// ---------------- K6: f1/f2 matvecs, layer 2 --------------------------------------
__global__ void k6_fvec2(const _Float16* __restrict__ h2T, const float* __restrict__ a_out,
                         float* __restrict__ f1, float* __restrict__ f2) {
  int tid = blockIdx.x * blockDim.x + threadIdx.x;
  if (tid >= BB * NN) return;
  int n = tid % NN;
  int b = tid / NN;
  const _Float16* hp = h2T + (size_t)b * DD * NN + n;
  float s1 = 0.f, s2 = 0.f;
  for (int e = 0; e < DD; ++e) {
    float hv = (float)hp[(size_t)e * NN];
    s1 += hv * a_out[e];
    s2 += hv * a_out[DD + e];
  }
  f1[tid] = s1; f2[tid] = s2;
}

// ---------------- K7: softmax, layer 2 (human rows only) --------------------------
__global__ void k7_att2(const float* __restrict__ f1, const float* __restrict__ f2,
                        _Float16* __restrict__ att, float* __restrict__ att_self) {
  __shared__ float red[128];
  int row = blockIdx.x;            // b*NH + n  (human)
  int t = threadIdx.x;
  int n = row % NH;
  int b = row / NH;
  float f1v = f1[(size_t)b * NN + n];
  const float* f2r = f2 + (size_t)b * NN + NH;   // objects
  float s[4];
  float lmax = -3.0e38f;
  #pragma unroll
  for (int i = 0; i < 4; ++i) {
    float x = f1v + f2r[t + i * 128];
    x = x > 0.f ? x : ALPHA * x;
    s[i] = x;
    lmax = fmaxf(lmax, x);
  }
  float xs = f1v + f2[(size_t)b * NN + n];
  float sself = xs > 0.f ? xs : ALPHA * xs;
  lmax = fmaxf(lmax, sself);
  red[t] = lmax; __syncthreads();
  for (int off = 64; off > 0; off >>= 1) { if (t < off) red[t] = fmaxf(red[t], red[t + off]); __syncthreads(); }
  float M = red[0]; __syncthreads();
  float lsum = 0.f;
  #pragma unroll
  for (int i = 0; i < 4; ++i) lsum += __expf(s[i] - M);
  if (t == 0) lsum += __expf(sself - M);
  red[t] = lsum; __syncthreads();
  for (int off = 64; off > 0; off >>= 1) { if (t < off) red[t] += red[t + off]; __syncthreads(); }
  float inv = 1.f / red[0];
  _Float16* arow = att + (size_t)row * 512;
  #pragma unroll
  for (int i = 0; i < 4; ++i) arow[t + i * 128] = (_Float16)(__expf(s[i] - M) * inv);
  if (t == 0) att_self[row] = __expf(sself - M) * inv;
}

// ---------------- K8: AV (layer 2), human rows only; 2x2 blocked ------------------
__global__ void k8_av2(const _Float16* __restrict__ h2T, const _Float16* __restrict__ att2,
                       float* __restrict__ out2) {
  int lane = threadIdx.x & 31;
  int wave = (blockIdx.x * blockDim.x + threadIdx.x) >> 5;
  int b = wave >> 6;               // 64 tiles of 32x32 per graph
  int r = wave & 63;
  int col0 = (r >> 2) << 5;        // n base (32 of 512 humans)
  int row0 = (r & 3) << 5;         // e base
  const _Float16* A  = h2T  + (size_t)b * DD * NN + NH;  // object columns
  const _Float16* Bt = att2 + (size_t)b * NH * 512;
  Acc4 acc = {};
  gemm_block_2x2(A, NN, row0, Bt, 512, col0, 512, lane, acc);
  int nlo = col0 + (lane & 15), nhi = nlo + 16;
  int elo = row0 + ((lane >> 4) << 3), ehi = elo + 16;
  float* base = out2 + (size_t)b * NH * DD;
  const v8f* accs[4] = { &acc.c00, &acc.c01, &acc.c10, &acc.c11 };
  int ns[4] = { nlo, nhi, nlo, nhi };
  int es[4] = { elo, elo, ehi, ehi };
  #pragma unroll
  for (int i = 0; i < 4; ++i) {
    const v8f& a = *accs[i];
    float* dst = base + (size_t)ns[i] * DD + es[i];
    *(float4*)dst       = make_float4(a[0], a[1], a[2], a[3]);
    *(float4*)(dst + 4) = make_float4(a[4], a[5], a[6], a[7]);
  }
}

// ---------------- K9: +self, elu, log_softmax, write output -----------------------
__global__ void k9_final(const float* __restrict__ out2, const float* __restrict__ as2,
                         const _Float16* __restrict__ h2T, float* __restrict__ out) {
  __shared__ float red[128];
  int row = blockIdx.x;            // b*NH + n
  int e = threadIdx.x;
  int b = row / NH;
  int n = row % NH;                // global node id == n
  float v = out2[(size_t)row * DD + e];
  v += as2[row] * (float)h2T[((size_t)b * DD + e) * NN + n];
  v = v > 0.f ? v : (__expf(v) - 1.f);
  red[e] = v; __syncthreads();
  for (int off = 64; off > 0; off >>= 1) { if (e < off) red[e] = fmaxf(red[e], red[e + off]); __syncthreads(); }
  float M = red[0]; __syncthreads();
  red[e] = __expf(v - M); __syncthreads();
  for (int off = 64; off > 0; off >>= 1) { if (e < off) red[e] += red[e + off]; __syncthreads(); }
  float L = __logf(red[0]);
  out[(size_t)row * DD + e] = v - M - L;
}

extern "C" void kernel_launch(void* const* d_in, const int* in_sizes, int n_in,
                              void* d_out, int out_size, void* d_ws, size_t ws_size,
                              hipStream_t stream) {
  (void)in_sizes; (void)n_in; (void)out_size; (void)ws_size;
  const float* h_states = (const float*)d_in[0];
  const float* o_states = (const float*)d_in[1];
  const float* W_heads  = (const float*)d_in[4];
  const float* a_heads  = (const float*)d_in[5];
  const float* W_out    = (const float*)d_in[6];
  const float* a_out    = (const float*)d_in[7];

  char* ws = (char*)d_ws;
  size_t off = 0;
  auto alloc = [&](size_t bytes) -> char* {
    off = (off + 255) & ~(size_t)255;
    char* p = ws + off;
    off += bytes;
    return p;
  };
  _Float16* xh   = (_Float16*)alloc((size_t)2 * BB * NN * DD);       // packed x
  _Float16* WtH  = (_Float16*)alloc((size_t)2 * HEADS * DD * DD);    // W_heads^T
  _Float16* Wt2  = (_Float16*)alloc((size_t)2 * DD * K2);            // W_out^T
  _Float16* h1T  = (_Float16*)alloc((size_t)2 * BB * HEADS * DD * NN);
  float*    f1a  = (float*)   alloc((size_t)4 * BB * HEADS * NN);
  float*    f2a  = (float*)   alloc((size_t)4 * BB * HEADS * NN);
  _Float16* att1 = (_Float16*)alloc((size_t)2 * BB * HEADS * NN * 512);
  float*    as1  = (float*)   alloc((size_t)4 * BB * HEADS * NN);
  _Float16* x2   = (_Float16*)alloc((size_t)2 * BB * NN * K2);
  _Float16* h2T  = (_Float16*)alloc((size_t)2 * BB * DD * NN);
  float*    f1b  = (float*)   alloc((size_t)4 * BB * NN);
  float*    f2b  = (float*)   alloc((size_t)4 * BB * NN);
  _Float16* att2 = (_Float16*)alloc((size_t)2 * BB * NH * 512);
  float*    as2  = (float*)   alloc((size_t)4 * BB * NH);
  float*    out2 = (float*)   alloc((size_t)4 * BB * NH * DD);

  const int prepN = BB * NN * DD + HEADS * DD * DD + DD * K2;
  k0_prep<<<(prepN + 255) / 256, 256, 0, stream>>>(h_states, o_states, W_heads, W_out,
                                                   xh, WtH, Wt2);
  k1_proj1<<<512, 256, 0, stream>>>(xh, WtH, h1T);                    // 4096 waves, 32x32/wave
  k2_fvec1<<<(BB * HEADS * NN + 255) / 256, 256, 0, stream>>>(h1T, a_heads, f1a, f2a);
  k3_att1 <<<BB * HEADS * NN, 128, 0, stream>>>(f1a, f2a, att1, as1);
  k4_av1  <<<512, 256, 0, stream>>>(h1T, att1, as1, x2);              // 4096 waves, 64 wmma each
  k5_proj2<<<128, 256, 0, stream>>>(x2, Wt2, h2T);                    // 1024 waves
  k6_fvec2<<<(BB * NN + 255) / 256, 256, 0, stream>>>(h2T, a_out, f1b, f2b);
  k7_att2 <<<BB * NH, 128, 0, stream>>>(f1b, f2b, att2, as2);
  k8_av2  <<<64, 256, 0, stream>>>(h2T, att2, out2);                  // 512 waves
  k9_final<<<BB * NH, 128, 0, stream>>>(out2, as2, h2T, (float*)d_out);
}
